// MapEncoder_24859270709593
// MI455X (gfx1250) — matrix operand order repaired and checked by reading, not compiled, not run on hardware
//
#include <hip/hip_runtime.h>

// ---------- problem constants ----------
static constexpr int kT = 4;
static constexpr int kE = 2048;
static constexpr int kP = 512;
static constexpr int kK = 5;
static constexpr int kDE = 16;     // D_ELEM
static constexpr int kH  = 128;    // D_HID == D_OUT
static constexpr int kMAP = 256;
static constexpr int kL1 = kP - kK + 1;       // 508
static constexpr int kL2 = kP - 2*kK + 2;     // 504
static constexpr int kROWS = 528;             // h1t rows: 0..507 data, 508..527 zero
static constexpr int kZROW = 520;             // dedicated zero row for K-pad

typedef _Float16 h16;
typedef __attribute__((ext_vector_type(8)))  _Float16 v8h;
typedef __attribute__((ext_vector_type(16))) _Float16 v16h;
typedef __attribute__((ext_vector_type(8)))  float    v8f;

__device__ __forceinline__ v16h cat8(v8h lo, v8h hi) {
    return __builtin_shufflevector(lo, hi, 0,1,2,3,4,5,6,7,8,9,10,11,12,13,14,15);
}

// LDS aperture maps flat addr[31:0] directly to the LDS byte offset (ISA 10.2),
// so truncating a generic pointer to a __shared__ object gives the LDS address.
__device__ __forceinline__ unsigned lds_addr_of(const void* p) {
    return (unsigned)(unsigned long long)p;
}

// Async copy 16B/lane from global to LDS (GLOBAL_LOAD_ASYNC_TO_LDS_B128,
// tracked by ASYNCcnt), then drain this wave's async queue.
__device__ __forceinline__ void async_copy_b128(unsigned lds_dst, const void* gsrc) {
    unsigned long long ga = (unsigned long long)gsrc;
    asm volatile("global_load_async_to_lds_b128 %0, %1, off"
                 :: "v"(lds_dst), "v"(ga) : "memory");
}
__device__ __forceinline__ void async_wait0() {
    asm volatile("s_wait_asynccnt 0" ::: "memory");
}

// ============================================================================
// Stage 1: per-element conv1 (scalar) -> conv2 (WMMA, im2col K=80 pad 96)
//          -> bias+ReLU -> sum over positions  => elem_latent [T,E,16]
// ============================================================================
__global__ __launch_bounds__(256) void conv_elem_kernel(
    const float* __restrict__ points,   // [T,E,P,2]
    const float* __restrict__ w1,       // [T,16,2,5]
    const float* __restrict__ b1,       // [T,16]
    const float* __restrict__ w2,       // [T,16,16,5]
    const float* __restrict__ b2,       // [T,16]
    float* __restrict__ elem_latent)    // [T,E,16]
{
    const int blk  = blockIdx.x;            // t*E + e
    const int t    = blk >> 11;             // E = 2048
    const int tid  = threadIdx.x;
    const int lane = tid & 31;
    const int wave = tid >> 5;

    __shared__ __align__(16) float ptsraw[kP * 2];      // 4 KB, interleaved x,y
    __shared__ __align__(32) h16 h1t[kROWS][kDE];       // 16.5 KB, transposed [p][ci]
    __shared__ __align__(32) h16 w2h[kDE][96];          // 3 KB, [co][kk], kk>=80 -> 0
    __shared__ float w1s[kDE * 2 * kK];                 // 160
    __shared__ float w2s[kDE * kDE * kK];               // 1280
    __shared__ float b1s[kDE], b2s[kDE];
    __shared__ float part[8][32][8];                    // 8 KB deterministic reduction

    // ---- async DMA of the 4KB points tile into LDS (1 x b128 per lane) ----
    const float* src = points + (size_t)blk * (kP * 2);
    __builtin_prefetch(src + kP * 2, 0, 1);   // global_prefetch_b8 (next element)
    async_copy_b128(lds_addr_of(&ptsraw[0]) + tid * 16, src + tid * 4);

    // ---- weights (small, regular loads); zero h1t (rows >= 508 stay zero) ----
    for (int i = tid; i < kDE*2*kK;   i += 256) w1s[i] = w1[t*kDE*2*kK + i];
    for (int i = tid; i < kDE*kDE*kK; i += 256) w2s[i] = w2[t*kDE*kDE*kK + i];
    if (tid < kDE) { b1s[tid] = b1[t*kDE + tid]; b2s[tid] = b2[t*kDE + tid]; }
    {
        v16h z = {};
        for (int r = tid; r < kROWS; r += 256) *(v16h*)&h1t[r][0] = z;
    }
    async_wait0();            // this wave's async deposit complete
    __syncthreads();          // all waves' deposits visible

    // ---- padded f16 copy of W2 in kk-major layout: w2h[co][kk], kk=k*16+ci ----
    for (int i = tid; i < kDE * 96; i += 256) {
        const int co = i / 96, kk = i - co * 96;
        float v = 0.0f;
        if (kk < 80) { const int ci = kk & 15, k = kk >> 4; v = w2s[co*80 + ci*5 + k]; }
        w2h[co][kk] = (h16)v;
    }

    // ---- conv1, thread-per-position: writes one contiguous 32B row of h1t ----
    for (int p = tid; p < kL1; p += 256) {
        float x0[kK], x1[kK];
        #pragma unroll
        for (int k = 0; k < kK; ++k) {
            x0[k] = ptsraw[2*(p + k)];
            x1[k] = ptsraw[2*(p + k) + 1];
        }
        v16h row;
        #pragma unroll
        for (int co = 0; co < kDE; ++co) {
            float acc = b1s[co];
            #pragma unroll
            for (int k = 0; k < kK; ++k) {
                acc += w1s[co*10 + k]     * x0[k];
                acc += w1s[co*10 + 5 + k] * x1[k];
            }
            row[co] = (h16)fmaxf(acc, 0.0f);
        }
        *(v16h*)&h1t[p][0] = row;
    }
    __syncthreads();

    // ---- conv2 WMMA: A = W2 [16 x 96], B = im2col patches (contiguous rows) ----
    const int arow = lane & 15;              // co
    const int koff = (lane >> 4) << 4;
    v16h afrag[3];
    #pragma unroll
    for (int kb = 0; kb < 3; ++kb) {
        v8h lo = *(const v8h*)&w2h[arow][kb*32 + koff];
        v8h hi = *(const v8h*)&w2h[arow][kb*32 + koff + 8];
        afrag[kb] = cat8(lo, hi);
    }

    float partial[8];
    #pragma unroll
    for (int r = 0; r < 8; ++r) partial[r] = 0.0f;
    const int ncol  = lane & 15;
    const int khalf = lane >> 4;             // B fragment K half select

    for (int tile = wave; tile < 32; tile += 8) {   // 32 tiles cover 504 (pad 512)
        const int pcol = tile*16 + ncol;            // output position for this lane
        v8f acc = {};
        #pragma unroll
        for (int kb = 0; kb < 3; ++kb) {
            const int k   = kb*2 + khalf;           // kbase/16 (uniform over halves)
            const int row = (k <= 4) ? (pcol + k) : kZROW;   // branch-free pad
            v16h bf = *(const v16h*)&h1t[row][0];   // 16 ci values, contiguous 32B
            acc = __builtin_amdgcn_wmma_f32_16x16x32_f16(
                false, afrag[kb], false, bf, (short)0, acc, false, false);
        }
        if (pcol < kL2) {
            #pragma unroll
            for (int r = 0; r < 8; ++r) {           // D: M = r + 8*(lane>>4)
                const int co = r + (khalf << 3);
                partial[r] += fmaxf(acc[r] + b2s[co], 0.0f);
            }
        }
    }
    #pragma unroll
    for (int r = 0; r < 8; ++r) part[wave][lane][r] = partial[r];
    __syncthreads();

    // deterministic reduction: elem_latent[co] = sum over waves & 16 lanes
    if (tid < kDE) {
        const int r  = tid & 7;
        const int lb = (tid >> 3) << 4;
        float s = 0.0f;
        for (int w = 0; w < 8; ++w)
            for (int l = 0; l < 16; ++l) s += part[w][lb + l][r];
        elem_latent[(size_t)blk * kDE + tid] = s;
    }
}

// ============================================================================
// Stage 2: r[t,:] = (sum_e X[t,e,:]) @ B[t]   (deterministic staged reduce)
// ============================================================================
template<int D>
__global__ __launch_bounds__(256) void reduce_r_kernel(
    const float* __restrict__ X,    // [T,E,D]
    const float* __restrict__ Bm,   // [T,D,H]
    float* __restrict__ rvec)       // [T,H]
{
    const int t = blockIdx.x, tid = threadIdx.x;
    __shared__ float lds[256];
    __shared__ float sD[D];
    constexpr int G = 256 / D;
    const int d = tid % D, g = tid / D;
    const float* Xt = X + (size_t)t * kE * D;
    float acc = 0.0f;
    for (int e = g; e < kE; e += G) acc += Xt[(size_t)e * D + d];
    lds[tid] = acc;
    __syncthreads();
    if (tid < D) {
        float s = 0.0f;
        for (int j = 0; j < G; ++j) s += lds[tid + j * D];
        sD[tid] = s;
    }
    __syncthreads();
    for (int h = tid; h < kH; h += 256) {
        float r = 0.0f;
        for (int dd = 0; dd < D; ++dd)
            r += sD[dd] * Bm[((size_t)t * D + dd) * kH + h];
        rvec[t * kH + h] = r;
    }
}

// ============================================================================
// Stage 3: Y = relu( X @ (A - B) + r )   [per type GEMM, M=2048, N=128]
// block owns one M-tile; wave w owns N-tile w; f16 LDS staging, coalesced fill
// ============================================================================
template<int DIN>
__global__ __launch_bounds__(256) void pn_gemm_kernel(
    const float* __restrict__ X,    // [T,E,DIN]
    const float* __restrict__ Am,   // [T,DIN,H]
    const float* __restrict__ Bm,   // [T,DIN,H]
    const float* __restrict__ rvec, // [T,H]
    float* __restrict__ Y)          // [T,E,H]
{
    constexpr int KPAD = (DIN + 31) & ~31;     // 32 or 128 (pow2)
    constexpr int KT   = KPAD / 32;
    const int blk   = blockIdx.x;              // t*(E/16) + mtile
    const int t     = blk / (kE / 16);
    const int mtile = blk % (kE / 16);
    const int tid = threadIdx.x, lane = tid & 31, wave = tid >> 5;

    __shared__ __align__(32) h16 xsh[16][KPAD];    // X tile, f16, K-padded
    __shared__ __align__(32) h16 wt[kH][KPAD];     // (A-B)^T : [n][kk]

    const float* Xt = X + ((size_t)t * kE + mtile * 16) * DIN;
    const float* At = Am + (size_t)t * DIN * kH;
    const float* Bt = Bm + (size_t)t * DIN * kH;

    // coalesced fill (kk fastest on global side); disjoint zero-pad pass
    for (int i = tid; i < 16 * DIN; i += 256) {
        const int r = i / DIN, c = i % DIN;        // DIN pow2
        xsh[r][c] = (h16)Xt[i];
    }
    for (int i = tid; i < 16 * (KPAD - DIN); i += 256) {
        const int r = i / (KPAD - DIN + (DIN == KPAD)), c = i % (KPAD - DIN + (DIN == KPAD));
        xsh[r][DIN + c] = (h16)0.0f;
    }
    for (int i = tid; i < DIN * kH; i += 256) {    // i = kk*128 + n  (coalesced)
        const int c = i >> 7, n = i & 127;
        wt[n][c] = (h16)(At[i] - Bt[i]);
    }
    for (int i = tid; i < (KPAD - DIN) * kH; i += 256) {
        const int c = i >> 7, n = i & 127;
        wt[n][DIN + c] = (h16)0.0f;
    }
    __syncthreads();

    const int arow = lane & 15;
    const int koff = (lane >> 4) << 4;
    const int ng   = wave * 16 + (lane & 15);      // global N
    const int kb16 = (lane >> 4) << 4;

    v8f acc = {};
    #pragma unroll
    for (int kb = 0; kb < KT; ++kb) {
        v8h alo = *(const v8h*)&xsh[arow][kb*32 + koff];
        v8h ahi = *(const v8h*)&xsh[arow][kb*32 + koff + 8];
        v16h af = cat8(alo, ahi);
        v16h bf = *(const v16h*)&wt[ng][kb*32 + kb16];
        acc = __builtin_amdgcn_wmma_f32_16x16x32_f16(
            false, af, false, bf, (short)0, acc, false, false);
    }
    const float rv = rvec[t * kH + ng];
    float* Yt = Y + ((size_t)t * kE + mtile * 16) * kH;
    #pragma unroll
    for (int r = 0; r < 8; ++r) {
        const int m = r + ((lane >> 4) << 3);
        Yt[(size_t)m * kH + ng] = fmaxf(acc[r] + rv, 0.0f);
    }
}

// ============================================================================
// Stage 4: max over E  -> M [T,128]
// ============================================================================
__global__ __launch_bounds__(256) void max_kernel(
    const float* __restrict__ G, float* __restrict__ M)
{
    const int t = blockIdx.x, tid = threadIdx.x;
    const int d = tid & 127, j = tid >> 7;
    __shared__ float lds[256];
    const float* Gt = G + (size_t)t * kE * kH;
    float m = -3.4e38f;
    for (int e = j; e < kE; e += 2) m = fmaxf(m, Gt[(size_t)e * kH + d]);
    lds[tid] = m;
    __syncthreads();
    if (tid < 128) M[t * kH + tid] = fmaxf(lds[tid], lds[tid + 128]);
}

// ============================================================================
// Stage 5: ot[t] = m[t] @ Aout[t] ; out = concat(ot) @ W_map + b_map
// ============================================================================
__global__ __launch_bounds__(256) void final_kernel(
    const float* __restrict__ M,     // [T,128]
    const float* __restrict__ Aout,  // [T,128,128]
    const float* __restrict__ Wmap,  // [512,256]
    const float* __restrict__ bmap,  // [256]
    float* __restrict__ out)         // [256]
{
    __shared__ float mb[kT * kH];
    __shared__ float lat[kT * kH];
    const int tid = threadIdx.x;
    for (int i = tid; i < kT * kH; i += 256) mb[i] = M[i];
    __syncthreads();
    for (int i = tid; i < kT * kH; i += 256) {
        const int t = i >> 7, o = i & 127;
        float s = 0.0f;
        for (int d = 0; d < kH; ++d)
            s += mb[t * kH + d] * Aout[((size_t)t * kH + d) * kH + o];
        lat[i] = s;
    }
    __syncthreads();
    float y = bmap[tid];
    for (int j = 0; j < kT * kH; ++j) y += lat[j] * Wmap[j * kMAP + tid];
    out[tid] = y;
}

// ============================================================================
extern "C" void kernel_launch(void* const* d_in, const int* in_sizes, int n_in,
                              void* d_out, int out_size, void* d_ws, size_t ws_size,
                              hipStream_t stream)
{
    (void)in_sizes; (void)n_in; (void)out_size; (void)ws_size;
    const float* points = (const float*)d_in[0];
    const float* w1     = (const float*)d_in[1];
    const float* b1     = (const float*)d_in[2];
    const float* w2     = (const float*)d_in[3];
    const float* b2     = (const float*)d_in[4];
    const float* A1     = (const float*)d_in[5];
    const float* B1     = (const float*)d_in[6];
    const float* A2     = (const float*)d_in[7];
    const float* B2     = (const float*)d_in[8];
    const float* Aout   = (const float*)d_in[9];
    const float* Wmap   = (const float*)d_in[10];
    const float* bmap   = (const float*)d_in[11];
    float* out = (float*)d_out;

    // workspace layout (all f32)
    char* ws = (char*)d_ws;
    size_t off = 0;
    float* elem = (float*)(ws + off); off += (size_t)kT * kE * kDE * 4;   // 512 KB
    float* r1   = (float*)(ws + off); off += 4096;                        // T*128
    float* g1   = (float*)(ws + off); off += (size_t)kT * kE * kH * 4;    // 4 MB
    float* r2   = (float*)(ws + off); off += 4096;
    float* g2   = (float*)(ws + off); off += (size_t)kT * kE * kH * 4;    // 4 MB
    float* mbuf = (float*)(ws + off); off += 4096;

    conv_elem_kernel<<<kT * kE, 256, 0, stream>>>(points, w1, b1, w2, b2, elem);
    reduce_r_kernel<kDE><<<kT, 256, 0, stream>>>(elem, B1, r1);
    pn_gemm_kernel<kDE><<<kT * (kE / 16), 256, 0, stream>>>(elem, A1, B1, r1, g1);
    reduce_r_kernel<kH><<<kT, 256, 0, stream>>>(g1, B2, r2);
    pn_gemm_kernel<kH><<<kT * (kE / 16), 256, 0, stream>>>(g1, A2, B2, r2, g2);
    max_kernel<<<kT, 256, 0, stream>>>(g2, mbuf);
    final_kernel<<<1, 256, 0, stream>>>(mbuf, Aout, Wmap, bmap, out);
}
